// HyperGRUUpdater_35321811042416
// MI455X (gfx1250) — matrix operand order repaired and checked by reading, not compiled, not run on hardware
//
#include <hip/hip_runtime.h>

// ---------------- types ----------------
typedef __bf16 v16bf __attribute__((ext_vector_type(16)));
typedef float  v8f   __attribute__((ext_vector_type(8)));
typedef unsigned int u32x4 __attribute__((ext_vector_type(4)));

union Frag { v16bf v; u32x4 q[2]; };

#define MIN_NORM 1e-15f
#define WAVE_BYTES 45056   // per-wave LDS partition (16B aligned)

// ---------------- helpers ----------------
__device__ __forceinline__ unsigned short f2bf(float f) {
  union { float f; unsigned u; } x; x.f = f;
  unsigned r = x.u + 0x7FFFu + ((x.u >> 16) & 1u);   // round-nearest-even
  return (unsigned short)(r >> 16);
}

__device__ __forceinline__ float artanh_(float x) {
  x = fminf(fmaxf(x, -1.0f + 1e-7f), 1.0f - 1e-7f);
  return 0.5f * (log1pf(x) - log1pf(-x));
}

// wave32: combine the two half-row partial sums (lanes r and r+16)
__device__ __forceinline__ float wsum(float s) { return s + __shfl_xor(s, 16, 32); }

// A fragment 16x32 bf16 from LDS (row-major, strideA elements).
// lane<16: row=lane, K = k0+0..7 (v0-3) and k0+16..23 (v4-7); lane>=16: K+8.
__device__ __forceinline__ Frag load_a(const unsigned short* lds, int lane, int k0, int strideA) {
  int row = lane & 15, half = lane >> 4;
  const unsigned short* p = lds + row * strideA + k0 + half * 8;
  Frag f;
  f.q[0] = *(const u32x4*)p;
  f.q[1] = *(const u32x4*)(p + 16);
  return f;
}

// B fragment 32x16 bf16 = Wᵀ slice; W row-major [128][K] bf16 in global.
// lane<16: N=lane, K = k0+0..7 (v0-3) and k0+16..23 (v4-7); lane>=16: N=lane-16, K+8.
__device__ __forceinline__ Frag load_b(const unsigned short* W, int lane, int n0, int k0, int K) {
  int n = n0 + (lane & 15);
  const unsigned short* p = W + n * K + k0 + ((lane >> 4) << 3);
  Frag f;
  f.q[0] = *(const u32x4*)p;
  f.q[1] = *(const u32x4*)(p + 16);
  return f;
}

// D (16x16 f32): lane<16: N=lane, v[r]=M=r; lane>=16: N=lane-16, v[r]=M=r+8.
__device__ __forceinline__ void store_d(float* out, int lane, int n0, v8f acc) {
  int n = n0 + (lane & 15), m0 = (lane >> 4) << 3;
#pragma unroll
  for (int i = 0; i < 8; ++i) out[(m0 + i) * 128 + n] = acc[i];
}

// out[16][128] f32 = A(16xK bf16, LDS) @ W(128xK bf16, global)ᵀ
// A fragments hoisted to registers once; inner loop streams only B from L2.
template <int K>
__device__ __forceinline__ void gemm_tile(const unsigned short* A, int strideA,
                                          const unsigned short* W, float* out, int lane) {
  Frag a[K / 32];
#pragma unroll
  for (int kt = 0; kt < K / 32; ++kt) a[kt] = load_a(A, lane, kt * 32, strideA);
  for (int nt = 0; nt < 8; ++nt) {
    v8f acc = {};
#pragma unroll
    for (int kt = 0; kt < K / 32; ++kt) {
      Frag b = load_b(W, lane, nt * 16, kt * 32, K);
      acc = __builtin_amdgcn_wmma_f32_16x16x32_bf16(false, a[kt].v, false, b.v,
                                                    (short)0, acc, false, false);
    }
    store_d(out, lane, nt * 16, acc);
  }
}

// mobius_matvec tail: B *= tanh(||B||/an * artanh(an)) / ||B|| per row
__device__ __forceinline__ void matvec_post(float* B, int lane, float an) {
  int row = lane & 15, fb = (lane >> 4) * 64;
  float* p = B + row * 128 + fb;
  float s = 0.f;
  for (int j = 0; j < 64; ++j) { float t = p[j]; s += t * t; }
  s = wsum(s);
  float mxn = fmaxf(sqrtf(s), MIN_NORM);
  float sc = tanhf(mxn / an * artanh_(an)) / mxn;
  for (int j = 0; j < 64; ++j) p[j] *= sc;
}

// D = mobius_add(X, Y) (c=1); D may alias X or Y
__device__ __forceinline__ void mobius_add_buf(const float* X, const float* Y, float* D, int lane) {
  int row = lane & 15, fb = (lane >> 4) * 64;
  const float* px = X + row * 128 + fb;
  const float* py = Y + row * 128 + fb;
  float x2 = 0.f, y2 = 0.f, xy = 0.f;
  for (int j = 0; j < 64; ++j) { float a = px[j], b = py[j]; x2 += a * a; y2 += b * b; xy += a * b; }
  x2 = wsum(x2); y2 = wsum(y2); xy = wsum(xy);
  float den = fmaxf(1.f + 2.f * xy + x2 * y2, MIN_NORM);
  float cx = (1.f + 2.f * xy + y2) / den, cy = (1.f - x2) / den;
  float* pd = D + row * 128 + fb;
  for (int j = 0; j < 64; ++j) pd[j] = cx * px[j] + cy * py[j];
}

// X = mobius_add(X, bvec) with broadcast 128-d bias vector
__device__ __forceinline__ void mobius_add_bias(float* X, const float* bvec, int lane) {
  int row = lane & 15, fb = (lane >> 4) * 64;
  float* px = X + row * 128 + fb;
  const float* pb = bvec + fb;
  float x2 = 0.f, y2 = 0.f, xy = 0.f;
  for (int j = 0; j < 64; ++j) { float a = px[j], b = pb[j]; x2 += a * a; y2 += b * b; xy += a * b; }
  x2 = wsum(x2); y2 = wsum(y2); xy = wsum(xy);
  float den = fmaxf(1.f + 2.f * xy + x2 * y2, MIN_NORM);
  float cx = (1.f + 2.f * xy + y2) / den, cy = (1.f - x2) / den;
  for (int j = 0; j < 64; ++j) px[j] = cx * px[j] + cy * pb[j];
}

// D = sigmoid(logmap0(X)); D may alias X
__device__ __forceinline__ void logmap0_sigmoid(const float* X, float* D, int lane) {
  int row = lane & 15, fb = (lane >> 4) * 64;
  const float* px = X + row * 128 + fb;
  float s = 0.f;
  for (int j = 0; j < 64; ++j) { float t = px[j]; s += t * t; }
  s = wsum(s);
  float n = fmaxf(sqrtf(s), MIN_NORM);
  float sc = artanh_(n) / n;
  float* pd = D + row * 128 + fb;
  for (int j = 0; j < 64; ++j) pd[j] = 1.f / (1.f + expf(-px[j] * sc));
}

// ---------------- prep: convert weights to bf16 in workspace ----------------
__global__ void prep_kernel(const float* __restrict__ wih, const float* __restrict__ whh,
                            const float* __restrict__ nw, unsigned short* __restrict__ ws) {
  int i = blockIdx.x * 256 + threadIdx.x;
  const int NIH = 384 * 256, NHH = 384 * 128, NNW = 128 * 128;
  if (i < NIH) {
    ws[i] = f2bf(wih[i]);
  } else if (i < NIH + NHH) {
    ws[i] = f2bf(whh[i - NIH]);
  } else if (i < NIH + NHH + NNW) {
    int j = i - NIH - NHH, r = j >> 7, c = j & 127;
    ws[i] = f2bf(c < 100 ? nw[r * 100 + c] : 0.f);   // pad node_W K:100->128
  }
}

// ---------------- main fused hyperbolic-GRU kernel ----------------
__global__ void __launch_bounds__(64)
hyper_gru_kernel(const float* __restrict__ mem_input, const float* __restrict__ mem,
                 const float* __restrict__ ts, const float* __restrict__ mem_ts,
                 const float* __restrict__ hfeat,
                 const float* __restrict__ time_w, const float* __restrict__ time_b,
                 const float* __restrict__ bias, const float* __restrict__ node_b,
                 const unsigned short* __restrict__ wsb,
                 float* __restrict__ out, int Ntot) {
  extern __shared__ char smem[];
  const int lane = threadIdx.x & 31;
  const int wave = threadIdx.x >> 5;
  char* wb = smem + wave * WAVE_BYTES;
  unsigned short* XB = (unsigned short*)wb;            // [16][256] bf16 x
  unsigned short* HB = (unsigned short*)(wb + 8192);   // [16][128] bf16 hx / rh / ...
  float* HXF = (float*)(wb + 12288);                   // [16][128] f32 hx
  float* B0  = (float*)(wb + 20480);                   // f32 scratch
  float* B1  = (float*)(wb + 28672);
  float* B2  = (float*)(wb + 36864);

  const unsigned short* WIH = wsb;                         // [384][256] (r,h,z)
  const unsigned short* WHH = wsb + 384 * 256;             // [384][128]
  const unsigned short* NW  = wsb + 384 * 256 + 384 * 128; // [128][128]

  const int row = lane & 15, half = lane >> 4;
  const int fb = half * 64;
  const int r0 = blockIdx.x * 32 + wave * 16;
  const int grow = min(r0 + row, Ntot - 1);

  // ---- phase 0: load x = [mem_input | proj(expmap0(time_feat))], hx; norms ----
  float dt = ts[grow] - mem_ts[grow];
  const float* mi = mem_input + (size_t)grow * 128;
  float sx = 0.f;
  for (int j = 0; j < 64; ++j) { float m = mi[fb + j]; XB[row * 256 + fb + j] = f2bf(m); sx += m * m; }
  float st = 0.f;
  for (int j = 0; j < 64; ++j) {
    int f = fb + j;
    float tf = cosf(dt * time_w[f] + time_b[f]);
    B0[row * 128 + f] = tf; st += tf * tf;
  }
  st = wsum(st);
  float tn = fmaxf(sqrtf(st), MIN_NORM);
  float en = tanhf(tn);               // ||expmap0(u)|| = tanh(||u||) for c=1
  float s1 = en / tn;
  const float maxn = 1.0f - 4e-3f;
  if (en > maxn) s1 *= maxn / en;     // proj back inside the ball
  float st2 = 0.f;
  for (int j = 0; j < 64; ++j) {
    float v = B0[row * 128 + fb + j] * s1;
    XB[row * 256 + 128 + fb + j] = f2bf(v);
    st2 += v * v;
  }
  sx = wsum(sx + st2);
  float xn = fmaxf(sqrtf(sx), MIN_NORM);

  const float* hxp = mem + (size_t)grow * 128;
  float sh = 0.f;
  for (int j = 0; j < 64; ++j) {
    float hv = hxp[fb + j];
    HXF[row * 128 + fb + j] = hv;
    HB[row * 128 + fb + j] = f2bf(hv);
    sh += hv * hv;
  }
  sh = wsum(sh);
  float hn = fmaxf(sqrtf(sh), MIN_NORM);

  // ---- z gate (chunk 2) ----
  gemm_tile<128>(HB, 128, WHH + 2 * 128 * 128, B0, lane);
  matvec_post(B0, lane, hn);
  gemm_tile<256>(XB, 256, WIH + 2 * 128 * 256, B1, lane);
  matvec_post(B1, lane, xn);
  mobius_add_buf(B0, B1, B1, lane);
  mobius_add_bias(B1, bias + 2 * 128, lane);
  logmap0_sigmoid(B1, B2, lane);               // z -> B2

  // ---- r gate (chunk 0) ----
  gemm_tile<128>(HB, 128, WHH, B0, lane);
  matvec_post(B0, lane, hn);
  gemm_tile<256>(XB, 256, WIH, B1, lane);
  matvec_post(B1, lane, xn);
  mobius_add_buf(B0, B1, B1, lane);
  mobius_add_bias(B1, bias, lane);
  logmap0_sigmoid(B1, B1, lane);               // r -> B1

  // ---- rh = mobius_pointwise_mul(r, hx); bf16 into HB ----
  float rhn;
  {
    float* pr = B1 + row * 128 + fb;
    float* ph = HXF + row * 128 + fb;
    float* pw = B0 + row * 128 + fb;
    float s = 0.f;
    for (int j = 0; j < 64; ++j) { float w = pr[j] * ph[j]; pw[j] = w; s += w * w; }
    s = wsum(s);
    float wxn = fmaxf(sqrtf(s), MIN_NORM);
    float t = tanhf(wxn / hn * artanh_(hn));
    float sc = t / wxn;
    unsigned short* pb = HB + row * 128 + fb;
    for (int j = 0; j < 64; ++j) pb[j] = f2bf(pw[j] * sc);
    rhn = fmaxf(t, MIN_NORM);                  // ||rh||
  }

  // ---- h_tilde (chunk 1) ----
  gemm_tile<128>(HB, 128, WHH + 128 * 128, B0, lane);
  matvec_post(B0, lane, rhn);
  gemm_tile<256>(XB, 256, WIH + 128 * 256, B1, lane);
  matvec_post(B1, lane, xn);
  mobius_add_buf(B0, B1, B1, lane);
  mobius_add_bias(B1, bias + 128, lane);       // h_tilde -> B1

  // ---- node-feature GEMM: reuse XB cols 0..127 as bf16 h (K padded to 128) ----
  {
    const float* hp = hfeat + (size_t)grow * 100;
    for (int j = 0; j < 64; ++j) {
      int f = fb + j;
      float v = (f < 100) ? hp[f] : 0.f;
      XB[row * 256 + f] = f2bf(v);
    }
  }
  gemm_tile<128>(XB, 256, NW, B0, lane);       // raw h @ node_Wᵀ -> B0

  // ---- delta = mobius_add(-hx, h_tilde) ----
  {
    float* pt = B1 + row * 128 + fb;
    float* ph = HXF + row * 128 + fb;
    float y2 = 0.f, xy = 0.f;
    for (int j = 0; j < 64; ++j) { float t = pt[j]; y2 += t * t; xy -= ph[j] * t; }
    y2 = wsum(y2); xy = wsum(xy);
    float den = fmaxf(1.f + 2.f * xy + sh * y2, MIN_NORM);
    float cx = (1.f + 2.f * xy + y2) / den, cy = (1.f - sh) / den;
    for (int j = 0; j < 64; ++j) pt[j] = cx * (-ph[j]) + cy * pt[j];
  }

  // ---- zd = mobius_pointwise_mul(z, delta) ----
  {
    float* pd = B1 + row * 128 + fb;
    float* pz = B2 + row * 128 + fb;
    float s = 0.f;
    for (int j = 0; j < 64; ++j) { float d = pd[j]; s += d * d; }
    s = wsum(s);
    float dn = fmaxf(sqrtf(s), MIN_NORM);
    float s2 = 0.f;
    for (int j = 0; j < 64; ++j) { float w = pz[j] * pd[j]; pd[j] = w; s2 += w * w; }
    s2 = wsum(s2);
    float wxn = fmaxf(sqrtf(s2), MIN_NORM);
    float sc = tanhf(wxn / dn * artanh_(dn)) / wxn;
    for (int j = 0; j < 64; ++j) pd[j] *= sc;
  }

  // ---- updated memory = mobius_add(hx, zd) ----
  mobius_add_buf(HXF, B1, B1, lane);

  // ---- h_out = mobius_add(updated, h@node_Wᵀ + node_b), store to global ----
  {
    float* pm = B0 + row * 128 + fb;
    for (int j = 0; j < 64; ++j) pm[j] += node_b[fb + j];
    float* pu = B1 + row * 128 + fb;
    float x2 = 0.f, y2 = 0.f, xy = 0.f;
    for (int j = 0; j < 64; ++j) { float a = pu[j], b = pm[j]; x2 += a * a; y2 += b * b; xy += a * b; }
    x2 = wsum(x2); y2 = wsum(y2); xy = wsum(xy);
    float den = fmaxf(1.f + 2.f * xy + x2 * y2, MIN_NORM);
    float cx = (1.f + 2.f * xy + y2) / den, cy = (1.f - x2) / den;
    if (r0 + row < Ntot) {
      float* po = out + (size_t)(r0 + row) * 128 + fb;
      for (int j = 0; j < 64; ++j) po[j] = cx * pu[j] + cy * pm[j];
    }
  }
}

// ---------------- host launcher ----------------
extern "C" void kernel_launch(void* const* d_in, const int* in_sizes, int n_in,
                              void* d_out, int out_size, void* d_ws, size_t ws_size,
                              hipStream_t stream) {
  (void)n_in; (void)out_size; (void)ws_size;
  const float* mem_input = (const float*)d_in[0];
  const float* mem       = (const float*)d_in[1];
  const float* ts        = (const float*)d_in[2];
  const float* mem_ts    = (const float*)d_in[3];
  const float* hfeat     = (const float*)d_in[4];
  const float* time_w    = (const float*)d_in[5];
  const float* time_b    = (const float*)d_in[6];
  const float* weight_ih = (const float*)d_in[7];
  const float* weight_hh = (const float*)d_in[8];
  const float* bias      = (const float*)d_in[9];
  const float* node_W    = (const float*)d_in[10];
  const float* node_b    = (const float*)d_in[11];
  float* out = (float*)d_out;
  unsigned short* wsb = (unsigned short*)d_ws;

  int N = in_sizes[0] / 128;

  const int prepElems = 384 * 256 + 384 * 128 + 128 * 128;
  prep_kernel<<<(prepElems + 255) / 256, 256, 0, stream>>>(weight_ih, weight_hh, node_W, wsb);

  int nBlocks = (N + 31) / 32;   // 32 rows per block (2 waves x 16-row tiles)
  hyper_gru_kernel<<<nBlocks, 64, 2 * WAVE_BYTES, stream>>>(
      mem_input, mem, ts, mem_ts, hfeat, time_w, time_b, bias, node_b, wsb, out, N);
}
